// VariationalGraohAutoEncoder_2044404433054
// MI455X (gfx1250) — matrix operand order repaired and checked by reading, not compiled, not run on hardware
//
#include <hip/hip_runtime.h>

// ---------------------------------------------------------------------------
// VGAE forward for MI455X (gfx1250, wave32, WMMA bf16 -> f32 accumulate)
// Round 3: templated (K, HAS_A2) GEMM -> fully unrolled k-loop, branch-free
// inner body, immediate-offset b128 fragment loads, batched load->wmma packs.
// ---------------------------------------------------------------------------

#define N_NODES 20000
#define N_EDGES 320000
#define IN_C    128
#define H_C     256
#define OUT_C   64

typedef __attribute__((ext_vector_type(16))) __bf16 v16bf;
typedef __attribute__((ext_vector_type(8)))  __bf16 v8bf;
typedef __attribute__((ext_vector_type(8)))  float  v8f;

union V16 { v16bf v; v8bf h[2]; };

// round-to-nearest-even fp32 -> bf16 bits
__device__ __forceinline__ unsigned short f2bf(float f) {
    union { float f; unsigned u; } c; c.f = f;
    unsigned u = c.u;
    u += 0x7fffu + ((u >> 16) & 1u);
    return (unsigned short)(u >> 16);
}
__device__ __forceinline__ float bf2f(unsigned short s) {
    union { unsigned u; float f; } c; c.u = ((unsigned)s) << 16;
    return c.f;
}

// ---------------------------------------------------------------------------
// Utility kernels
// ---------------------------------------------------------------------------
__global__ void zero_k(float* __restrict__ p, long long n) {
    long long i = (long long)blockIdx.x * blockDim.x + threadIdx.x;
    if (i < n) p[i] = 0.0f;
}

// fp32 -> bf16, coalesced
__global__ void cvt_k(const float* __restrict__ src, unsigned short* __restrict__ dst,
                      long long n) {
    long long i = (long long)blockIdx.x * blockDim.x + threadIdx.x;
    if (i < n) dst[i] = f2bf(src[i]);
}

// repack weight K x N fp32 (row major) -> N x K bf16 (transposed)
__global__ void repack_w_k(const float* __restrict__ W, unsigned short* __restrict__ Wt,
                           int K, int N) {
    int idx = blockIdx.x * blockDim.x + threadIdx.x;
    if (idx >= K * N) return;
    int k = idx / N, n = idx % N;
    Wt[(long long)n * K + k] = f2bf(W[idx]);
}

__global__ void degree_k(const int* __restrict__ ei, float* __restrict__ deg) {
    int e = blockIdx.x * blockDim.x + threadIdx.x;
    if (e < N_EDGES) atomicAdd(&deg[ei[N_EDGES + e]], 1.0f);
}

// one block per edge, blockDim == C: bf16 gather (half traffic) + fp32 atomic add
__global__ void scatter_add_k(const unsigned short* __restrict__ xin,
                              const int* __restrict__ ei,
                              float* __restrict__ agg, int C) {
    int e = blockIdx.x;
    int s = ei[e];
    int d = ei[N_EDGES + e];
    float v = bf2f(xin[(long long)s * C + threadIdx.x]);
    atomicAdd(&agg[(long long)d * C + threadIdx.x], v);
}

// agg / max(deg,1) -> bf16
__global__ void normalize_cvt_k(const float* __restrict__ agg, const float* __restrict__ deg,
                                unsigned short* __restrict__ out, int C) {
    int i = blockIdx.x;
    float r = 1.0f / fmaxf(deg[i], 1.0f);
    long long idx = (long long)i * C + threadIdx.x;
    out[idx] = f2bf(agg[idx] * r);
}

// ---------------------------------------------------------------------------
// Dual-A WMMA GEMM:  D = [relu?]( A1@W1 + (HAS_A2 ? A2@W2 : 0) + bias ) [+ res]
//   A1,A2 : M x K bf16 row-major        W1t,W2t : N x K bf16 (pre-transposed)
//   res   : M x N fp32 (nullable)       outputs: outf fp32 / outb bf16 (nullable)
//   M%16==0, N%64==0, K in {128,256} (template, fully unrolled).
// One wave32 owns a 16x64 strip (4 n-tiles). Per k-step: load A frags,
// load all B frags (packed v8bf -> global_load_b128 with immediate offsets),
// then issue the 4/8 WMMAs back-to-back. Branch-free inner body; all control
// flow wave-uniform => EXEC all-1s at every WMMA (ISA 7.12 requirement).
// Fragment layouts per ISA 7.12.2:
//   A lane (m = lane&15, half = lane>>4): elems 0-7 -> K = k0+8*half+0..7,
//                                         elems 8-15 -> K = k0+16+8*half+0..7
//   B lane (n = lane&15, half): elems 0-7 -> K = k0+8*half+0..7, 8-15 -> K = k0+16+8*half+0..7
//   C/D: VGPR r -> M = 8*half + r, N = lane&15
// ---------------------------------------------------------------------------
template<int K, bool HAS_A2>
__global__ __launch_bounds__(256)
void sage_gemm_k(const __bf16* __restrict__ A1, const __bf16* __restrict__ W1t,
                 const __bf16* __restrict__ A2, const __bf16* __restrict__ W2t,
                 const float* __restrict__ bias, const float* __restrict__ res,
                 float* __restrict__ outf, unsigned short* __restrict__ outb,
                 int M, int N, int relu) {
    const int wave = (blockIdx.x * blockDim.x + threadIdx.x) >> 5;
    const int lane = threadIdx.x & 31;
    const int RT = M >> 4;          // 16-row tiles
    const int CG = N >> 6;          // 64-col groups
    if (wave >= RT * CG) return;    // wave-uniform
    const int rt = wave / CG;
    const int cg = wave % CG;

    const int half = lane >> 4;     // 0: lanes 0-15, 1: lanes 16-31
    const int nlo  = lane & 15;

    const __bf16* a1p = A1 + (long long)(rt * 16 + nlo) * K + half * 8;
    const __bf16* a2p = HAS_A2 ? (A2 + (long long)(rt * 16 + nlo) * K + half * 8) : nullptr;

    const __bf16* w1n[4];
    const __bf16* w2n[4];
    #pragma unroll
    for (int nt = 0; nt < 4; ++nt) {
        const int n = cg * 64 + nt * 16 + nlo;
        w1n[nt] = W1t + (long long)n * K + half * 8;
        if constexpr (HAS_A2) w2n[nt] = W2t + (long long)n * K + half * 8;
    }

    const v8f vzero = {0.f, 0.f, 0.f, 0.f, 0.f, 0.f, 0.f, 0.f};
    v8f acc[4];
    #pragma unroll
    for (int t = 0; t < 4; ++t) acc[t] = vzero;

    #pragma unroll
    for (int k0 = 0; k0 < K; k0 += 32) {
        // ---- load pack: A fragments ----
        V16 a1, a2;
        a1.h[0] = *(const v8bf*)(a1p + k0);
        a1.h[1] = *(const v8bf*)(a1p + k0 + 16);
        if constexpr (HAS_A2) {
            a2.h[0] = *(const v8bf*)(a2p + k0);
            a2.h[1] = *(const v8bf*)(a2p + k0 + 16);
        }
        // ---- load pack: all B fragments for this k-step ----
        V16 b[4], b2[4];
        #pragma unroll
        for (int nt = 0; nt < 4; ++nt) {
            b[nt].h[0] = *(const v8bf*)(w1n[nt] + k0);
            b[nt].h[1] = *(const v8bf*)(w1n[nt] + k0 + 16);
            if constexpr (HAS_A2) {
                b2[nt].h[0] = *(const v8bf*)(w2n[nt] + k0);
                b2[nt].h[1] = *(const v8bf*)(w2n[nt] + k0 + 16);
            }
        }
        // ---- compute pack: 4 or 8 WMMAs back-to-back ----
        #pragma unroll
        for (int nt = 0; nt < 4; ++nt) {
            acc[nt] = __builtin_amdgcn_wmma_f32_16x16x32_bf16(
                false, a1.v, false, b[nt].v, (short)0, acc[nt], false, false);
            if constexpr (HAS_A2) {
                acc[nt] = __builtin_amdgcn_wmma_f32_16x16x32_bf16(
                    false, a2.v, false, b2[nt].v, (short)0, acc[nt], false, false);
            }
        }
    }

    // Epilogue: C/D layout -> VGPR r holds M = rt*16 + 8*half + r, N = n
    #pragma unroll
    for (int nt = 0; nt < 4; ++nt) {
        const int n = cg * 64 + nt * 16 + nlo;
        const float bv = bias ? bias[n] : 0.0f;
        #pragma unroll
        for (int r = 0; r < 8; ++r) {
            const long long m = rt * 16 + half * 8 + r;
            float v = acc[nt][r] + bv;
            if (relu) v = fmaxf(v, 0.0f);
            if (res)  v += res[m * N + n];
            if (outf) outf[m * N + n] = v;
            if (outb) outb[m * N + n] = f2bf(v);
        }
    }
}

// ---------------------------------------------------------------------------
// Host-side orchestration
// ---------------------------------------------------------------------------
extern "C" void kernel_launch(void* const* d_in, const int* in_sizes, int n_in,
                              void* d_out, int out_size, void* d_ws, size_t ws_size,
                              hipStream_t stream) {
    (void)in_sizes; (void)n_in; (void)out_size; (void)ws_size;

    const float* x    = (const float*)d_in[0];
    const int*   ei   = (const int*)d_in[1];     // [2, E] flat: row0=src, row1=dst
    const float* W1l  = (const float*)d_in[2];
    const float* b1   = (const float*)d_in[3];
    const float* W1r  = (const float*)d_in[4];
    const float* W2l  = (const float*)d_in[5];
    const float* b2   = (const float*)d_in[6];
    const float* W2r  = (const float*)d_in[7];
    const float* Wres = (const float*)d_in[8];
    const float* bres = (const float*)d_in[9];
    const float* Wmul = (const float*)d_in[10];
    const float* bmu  = (const float*)d_in[11];
    const float* Wmur = (const float*)d_in[12];
    const float* Wlsl = (const float*)d_in[13];
    const float* bls  = (const float*)d_in[14];
    const float* Wlsr = (const float*)d_in[15];

    float* outp   = (float*)d_out;
    float* mu     = outp;                        // [N, 64]
    float* logstd = outp + (long long)N_NODES * OUT_C;

    // ---- workspace carve-up (~78 MB; activations stay L2-resident) ----
    const long long NN128 = (long long)N_NODES * IN_C;   // 2,560,000
    const long long NN256 = (long long)N_NODES * H_C;    // 5,120,000

    float* wsf  = (float*)d_ws;
    float* deg  = wsf;                       // 20480 floats (padded)
    float* aggf = deg  + 20480;              // NN256 floats (reused per stage)
    float* xres = aggf + NN256;              // NN256 floats

    unsigned short* wsb = (unsigned short*)(xres + NN256);
    unsigned short* xb   = wsb;              // NN128 bf16
    unsigned short* aggb = xb   + NN128;     // NN256 bf16 (reused per stage)
    unsigned short* h1b  = aggb + NN256;     // NN256 bf16
    unsigned short* h2b  = h1b  + NN256;     // NN256 bf16
    // transposed bf16 weights
    unsigned short* W1lt = h2b  + NN256;     // 128*256
    unsigned short* W1rt = W1lt + IN_C * H_C;
    unsigned short* Wrst = W1rt + IN_C * H_C;
    unsigned short* W2lt = Wrst + IN_C * H_C;
    unsigned short* W2rt = W2lt + H_C * H_C;
    unsigned short* Wmlt = W2rt + H_C * H_C;
    unsigned short* Wmrt = Wmlt + H_C * OUT_C;
    unsigned short* Wllt = Wmrt + H_C * OUT_C;
    unsigned short* Wlrt = Wllt + H_C * OUT_C;

    // ---- one-time conversions (bf16 activations + transposed bf16 weights) ----
    cvt_k<<<(unsigned)((NN128 + 255) / 256), 256, 0, stream>>>(x, xb, NN128);
    repack_w_k<<<(IN_C * H_C + 255) / 256, 256, 0, stream>>>(W1l, W1lt, IN_C, H_C);
    repack_w_k<<<(IN_C * H_C + 255) / 256, 256, 0, stream>>>(W1r, W1rt, IN_C, H_C);
    repack_w_k<<<(IN_C * H_C + 255) / 256, 256, 0, stream>>>(Wres, Wrst, IN_C, H_C);
    repack_w_k<<<(H_C * H_C + 255) / 256, 256, 0, stream>>>(W2l, W2lt, H_C, H_C);
    repack_w_k<<<(H_C * H_C + 255) / 256, 256, 0, stream>>>(W2r, W2rt, H_C, H_C);
    repack_w_k<<<(H_C * OUT_C + 255) / 256, 256, 0, stream>>>(Wmul, Wmlt, H_C, OUT_C);
    repack_w_k<<<(H_C * OUT_C + 255) / 256, 256, 0, stream>>>(Wmur, Wmrt, H_C, OUT_C);
    repack_w_k<<<(H_C * OUT_C + 255) / 256, 256, 0, stream>>>(Wlsl, Wllt, H_C, OUT_C);
    repack_w_k<<<(H_C * OUT_C + 255) / 256, 256, 0, stream>>>(Wlsr, Wlrt, H_C, OUT_C);

    // ---- degree (shared by all stages) ----
    zero_k<<<(20480 + 255) / 256, 256, 0, stream>>>(deg, 20480);
    degree_k<<<(N_EDGES + 255) / 256, 256, 0, stream>>>(ei, deg);

    // ---- stage 1: h1 = relu( mean_agg(x)@W1l + x@W1r + b1 ) -> bf16 ----
    zero_k<<<(unsigned)((NN128 + 255) / 256), 256, 0, stream>>>(aggf, NN128);
    scatter_add_k<<<N_EDGES, IN_C, 0, stream>>>(xb, ei, aggf, IN_C);
    normalize_cvt_k<<<N_NODES, IN_C, 0, stream>>>(aggf, deg, aggb, IN_C);
    // M=20000 K=128 N=256 -> 1250*4 = 5000 waves -> 625 blocks
    sage_gemm_k<128, true><<<625, 256, 0, stream>>>(
        (const __bf16*)aggb, (const __bf16*)W1lt, (const __bf16*)xb, (const __bf16*)W1rt,
        b1, nullptr, nullptr, h1b, N_NODES, H_C, /*relu=*/1);

    // ---- residual: xres = x@Wres + bres (fp32, consumed by stage-2 epilogue) ----
    sage_gemm_k<128, false><<<625, 256, 0, stream>>>(
        (const __bf16*)xb, (const __bf16*)Wrst, nullptr, nullptr,
        bres, nullptr, xres, nullptr, N_NODES, H_C, /*relu=*/0);

    // ---- stage 2: h2 = relu( mean_agg(h1)@W2l + h1@W2r + b2 ) + xres -> bf16 ----
    zero_k<<<(unsigned)((NN256 + 255) / 256), 256, 0, stream>>>(aggf, NN256);
    scatter_add_k<<<N_EDGES, H_C, 0, stream>>>(h1b, ei, aggf, H_C);
    normalize_cvt_k<<<N_NODES, H_C, 0, stream>>>(aggf, deg, aggb, H_C);
    sage_gemm_k<256, true><<<625, 256, 0, stream>>>(
        (const __bf16*)aggb, (const __bf16*)W2lt, (const __bf16*)h1b, (const __bf16*)W2rt,
        b2, xres, nullptr, h2b, N_NODES, H_C, /*relu=*/1);

    // ---- stage 3: shared agg(h2); mu & logstd heads (fp32 to d_out) ----
    zero_k<<<(unsigned)((NN256 + 255) / 256), 256, 0, stream>>>(aggf, NN256);
    scatter_add_k<<<N_EDGES, H_C, 0, stream>>>(h2b, ei, aggf, H_C);
    normalize_cvt_k<<<N_NODES, H_C, 0, stream>>>(aggf, deg, aggb, H_C);
    // M=20000 K=256 N=64 -> 1250 waves -> 157 blocks
    sage_gemm_k<256, true><<<157, 256, 0, stream>>>(
        (const __bf16*)aggb, (const __bf16*)Wmlt, (const __bf16*)h2b, (const __bf16*)Wmrt,
        bmu, nullptr, mu, nullptr, N_NODES, OUT_C, /*relu=*/0);
    sage_gemm_k<256, true><<<157, 256, 0, stream>>>(
        (const __bf16*)aggb, (const __bf16*)Wllt, (const __bf16*)h2b, (const __bf16*)Wlrt,
        bls, nullptr, logstd, nullptr, N_NODES, OUT_C, /*relu=*/0);
}
